// Decoder_19189913878746
// MI455X (gfx1250) — compile-verified, matrix operand order checked by reading, not run on hardware
//
#include <hip/hip_runtime.h>
#include <hip/hip_bf16.h>
#include <math.h>

typedef __attribute__((ext_vector_type(16))) _Float16 v16h;
typedef __attribute__((ext_vector_type(8)))  float    v8f;
typedef __attribute__((ext_vector_type(4)))  int      v4i;

// ---------------- constants ----------------
static constexpr int BATCH = 1024;
static constexpr int LAT   = 1024;
static constexpr int HID   = 1024;
static constexpr int TT    = 32;

// GEMM tiling
static constexpr int BN = 128;
static constexpr int BK = 32;
static constexpr int LDS_STRIDE = BK + 8;   // halfs; pad to dodge bank conflicts

// ---------------------------------------------------------------------------
// CDNA5 async global->LDS staging (ASYNCcnt-tracked, no VGPR round-trip).
// Builtin takes typed v4i pointers: (v4i AS1* gsrc, v4i AS3* ldst, off, cpol).
// Guarded: falls back to a sync copy if the builtin is absent.
// ---------------------------------------------------------------------------
#if defined(__has_builtin)
#if __has_builtin(__builtin_amdgcn_global_load_async_to_lds_b128)
#define USE_ASYNC_LDS 1
#endif
#endif

typedef __attribute__((address_space(1))) v4i* gas_v4i_p;
typedef __attribute__((address_space(3))) v4i* las_v4i_p;

__device__ __forceinline__ void cp16_to_lds(const _Float16* g, _Float16* l) {
#ifdef USE_ASYNC_LDS
    __builtin_amdgcn_global_load_async_to_lds_b128(
        (gas_v4i_p)(v4i*)(g), (las_v4i_p)(v4i*)(l), 0, 0);
#else
    *(uint4*)l = *(const uint4*)g;
#endif
}

__device__ __forceinline__ void async_wait0() {
#ifdef USE_ASYNC_LDS
#if __has_builtin(__builtin_amdgcn_s_wait_asynccnt)
    __builtin_amdgcn_s_wait_asynccnt(0);
#else
    asm volatile("s_wait_asynccnt 0" ::: "memory");
#endif
#endif
}

__device__ __forceinline__ v16h load_frag(const _Float16* p) {
    union { v16h v; uint4 q[2]; } u;
    u.q[0] = *(const uint4*)(p);        // K = hi*8 .. +7
    u.q[1] = *(const uint4*)(p + 16);   // K = hi*8+16 .. +23
    return u.v;
}

// ---------------------------------------------------------------------------
// Tiled WMMA GEMM (f16 inputs, f32 accumulate), double-buffered async LDS:
//   C[M,N] = init + A1[M,K] * B1t[N,K]^T (+ A2 * B2t^T)   (optional tanh)
// init = Cinit[M,N] if non-null, else bias[N] broadcast, else 0.
// BMT = 16-row M-tiles per wave (block tile = BMT*16 x 128), 8 waves,
// each wave owns a 16-wide N strip and runs BMT back-to-back v_wmma.
// ---------------------------------------------------------------------------
template<int BMT>
__global__ __launch_bounds__(256)
void gemm_f16_wmma(const _Float16* __restrict__ A1, const _Float16* __restrict__ B1t,
                   const _Float16* __restrict__ A2, const _Float16* __restrict__ B2t,
                   const float* __restrict__ Cinit, const float* __restrict__ bias,
                   float* __restrict__ C, int M, int N, int K, int do_tanh)
{
    constexpr int BM_ = BMT * 16;
    constexpr int ASZ = BM_ * LDS_STRIDE;
    constexpr int BSZ = BN  * LDS_STRIDE;
    __shared__ _Float16 sA[2 * ASZ];
    __shared__ _Float16 sB[2 * BSZ];

    const int tid  = threadIdx.x;
    const int wave = tid >> 5;        // 0..7: 16-wide N strip
    const int lane = tid & 31;
    const int lr   = lane & 15;
    const int hi   = lane >> 4;       // K-half selector (A/B), M-half selector (C/D)
    const int m0   = blockIdx.y * BM_;
    const int n0   = blockIdx.x * BN;
    const int col  = n0 + wave * 16 + lr;

    v8f acc[BMT];
    #pragma unroll
    for (int mt = 0; mt < BMT; ++mt) {
        #pragma unroll
        for (int r = 0; r < 8; ++r) {
            const int row = m0 + mt * 16 + r + hi * 8;
            float v = 0.f;
            if (Cinit)      v = Cinit[(size_t)row * N + col];
            else if (bias)  v = bias[col];
            acc[mt][r] = v;
        }
    }

    const int kpb    = K / BK;                 // K-steps per source pair
    const int nsteps = (A2 ? 2 : 1) * kpb;

    auto issue = [&](int step, int buf) {
        const _Float16* __restrict__ Asrc = (step < kpb) ? A1 : A2;
        const _Float16* __restrict__ Bsrc = (step < kpb) ? B1t : B2t;
        const int k0 = ((step < kpb) ? step : step - kpb) * BK;
        _Float16* dA = sA + buf * ASZ;
        _Float16* dB = sB + buf * BSZ;
        #pragma unroll
        for (int i = 0; i < BM_ / 64; ++i) {   // BM_ x 32 halfs = BM_*4 16B chunks
            const int idx = tid + i * 256;
            const int r = idx >> 2, ch = idx & 3;
            cp16_to_lds(Asrc + (size_t)(m0 + r) * K + k0 + ch * 8,
                        &dA[r * LDS_STRIDE + ch * 8]);
        }
        #pragma unroll
        for (int i = 0; i < BN / 64; ++i) {    // 128 x 32 halfs = 512 chunks
            const int idx = tid + i * 256;
            const int r = idx >> 2, ch = idx & 3;
            cp16_to_lds(Bsrc + (size_t)(n0 + r) * K + k0 + ch * 8,
                        &dB[r * LDS_STRIDE + ch * 8]);
        }
    };

    issue(0, 0);
    int p = 0;
    for (int step = 0; step < nsteps; ++step) {
        async_wait0();          // this wave's copies into buf p have landed
        __syncthreads();        // everyone's copies visible; prior reads retired
        if (step + 1 < nsteps) issue(step + 1, p ^ 1);   // overlap next tile

        const _Float16* bufA = sA + p * ASZ;
        const _Float16* bufB = sB + p * BSZ;

        const v16h bf = load_frag(&bufB[(wave * 16 + lr) * LDS_STRIDE + hi * 8]);
        v16h af[BMT];
        #pragma unroll
        for (int mt = 0; mt < BMT; ++mt)
            af[mt] = load_frag(&bufA[(mt * 16 + lr) * LDS_STRIDE + hi * 8]);
        #pragma unroll
        for (int mt = 0; mt < BMT; ++mt)
            acc[mt] = __builtin_amdgcn_wmma_f32_16x16x32_f16(
                false, af[mt], false, bf, (short)0, acc[mt], false, false);
        p ^= 1;
    }

    #pragma unroll
    for (int mt = 0; mt < BMT; ++mt) {
        #pragma unroll
        for (int r = 0; r < 8; ++r) {
            const int row = m0 + mt * 16 + r + hi * 8;
            float v = acc[mt][r];
            if (do_tanh) v = tanhf(v);
            C[(size_t)row * N + col] = v;
        }
    }
}

// ---------------- prep kernels ----------------
__global__ void cvt_f16_kernel(const float* __restrict__ src, _Float16* __restrict__ dst, int n) {
    int i = blockIdx.x * blockDim.x + threadIdx.x;
    if (i < n) dst[i] = (_Float16)src[i];
}
__global__ void copy_f32_kernel(const float* __restrict__ src, float* __restrict__ dst, int n) {
    int i = blockIdx.x * blockDim.x + threadIdx.x;
    if (i < n) dst[i] = src[i];
}
__global__ void add_vec_kernel(const float* __restrict__ a, const float* __restrict__ b,
                               float* __restrict__ dst, int n) {
    int i = blockIdx.x * blockDim.x + threadIdx.x;
    if (i < n) dst[i] = a[i] + b[i];
}
// src: (K x N) f32 row-major -> dst: (N x K) f16 row-major
__global__ void transpose_cvt_kernel(const float* __restrict__ src, _Float16* __restrict__ dst,
                                     int K, int N) {
    __shared__ float tile[32][33];
    const int kb = blockIdx.y * 32, nb = blockIdx.x * 32;
    const int tx = threadIdx.x, ty = threadIdx.y;   // (32, 8)
    #pragma unroll
    for (int i = 0; i < 32; i += 8)
        tile[ty + i][tx] = src[(size_t)(kb + ty + i) * N + nb + tx];
    __syncthreads();
    #pragma unroll
    for (int i = 0; i < 32; i += 8)
        dst[(size_t)(nb + ty + i) * K + kb + tx] = (_Float16)tile[tx][ty + i];
}

// ---------------- per-step kernels ----------------
__global__ void lstm_gates_kernel(const float* __restrict__ z, float* __restrict__ c,
                                  float* __restrict__ h, _Float16* __restrict__ c_h,
                                  _Float16* __restrict__ h_h, int H) {
    const int idx = blockIdx.x * blockDim.x + threadIdx.x;
    const int b = idx / H, j = idx - b * H;
    const size_t base = (size_t)b * 4 * H;
    const float zi = z[base + j];
    const float zf = z[base + H + j];
    const float zg = z[base + 2 * H + j];
    const float zo = z[base + 3 * H + j];
    const float si = 1.f / (1.f + expf(-zi));
    const float sf = 1.f / (1.f + expf(-zf));
    const float so = 1.f / (1.f + expf(-zo));
    const float cn = sf * c[idx] + si * tanhf(zg);
    const float hn = so * tanhf(cn);
    c[idx] = cn;  h[idx] = hn;
    c_h[idx] = (_Float16)cn;  h_h[idx] = (_Float16)hn;
}

// score already tanh'ed by GEMM epilogue; softmax over feature axis, ctx = beta*latent
__global__ __launch_bounds__(256)
void softmax_ctx_kernel(const float* __restrict__ score, const float* __restrict__ latent,
                        _Float16* __restrict__ ctx_h, int D) {
    __shared__ float red[256];
    const int b = blockIdx.x, tid = threadIdx.x;
    const float* s = score + (size_t)b * D;
    float mx = -3.4e38f;
    for (int j = tid; j < D; j += 256) mx = fmaxf(mx, s[j]);
    red[tid] = mx; __syncthreads();
    for (int st = 128; st > 0; st >>= 1) { if (tid < st) red[tid] = fmaxf(red[tid], red[tid + st]); __syncthreads(); }
    mx = red[0]; __syncthreads();
    float sum = 0.f;
    for (int j = tid; j < D; j += 256) sum += expf(s[j] - mx);
    red[tid] = sum; __syncthreads();
    for (int st = 128; st > 0; st >>= 1) { if (tid < st) red[tid] += red[tid + st]; __syncthreads(); }
    const float inv = 1.f / red[0];
    for (int j = tid; j < D; j += 256)
        ctx_h[(size_t)b * D + j] = (_Float16)(expf(s[j] - mx) * inv * latent[(size_t)b * D + j]);
}

__global__ __launch_bounds__(256)
void out_mu_kernel(const float* __restrict__ h, const float* __restrict__ Wmu,
                   const float* __restrict__ bmu, float* __restrict__ out,
                   int t, int T, int H) {
    __shared__ float red[256];
    const int b = blockIdx.x, tid = threadIdx.x;
    float sum = 0.f;
    for (int j = tid; j < H; j += 256) sum += h[(size_t)b * H + j] * Wmu[j];
    red[tid] = sum; __syncthreads();
    for (int st = 128; st > 0; st >>= 1) { if (tid < st) red[tid] += red[tid + st]; __syncthreads(); }
    if (tid == 0) out[(size_t)b * T + t] = red[0] + bmu[0];
}

// ---------------------------------------------------------------------------
extern "C" void kernel_launch(void* const* d_in, const int* in_sizes, int n_in,
                              void* d_out, int out_size, void* d_ws, size_t ws_size,
                              hipStream_t stream) {
    const float* latent = (const float*)d_in[0];
    const float* h0     = (const float*)d_in[1];
    const float* c0     = (const float*)d_in[2];
    const float* Wk     = (const float*)d_in[3];   // (2L, 4H)
    const float* Wr     = (const float*)d_in[4];   // (H, 4H)
    const float* bb     = (const float*)d_in[5];   // (4H)
    const float* W1     = (const float*)d_in[6];   // (L, L)
    const float* b1     = (const float*)d_in[7];
    const float* W2     = (const float*)d_in[8];   // (2H, L)
    const float* b2     = (const float*)d_in[9];
    const float* Wmu    = (const float*)d_in[10];  // (H, 1)
    const float* bmu    = (const float*)d_in[11];
    float* out          = (float*)d_out;

    const int B = BATCH, L = LAT, H = HID, T = TT;
    const int H4 = 4 * H;

    // ---- carve workspace ----
    char* w = (char*)d_ws;
    auto alloc = [&](size_t bytes) { void* p = (void*)w; w += (bytes + 255) & ~(size_t)255; return p; };
    _Float16* lat_h   = (_Float16*)alloc((size_t)B * L * 2);
    _Float16* h_h     = (_Float16*)alloc((size_t)B * H * 2);
    _Float16* c_h     = (_Float16*)alloc((size_t)B * H * 2);
    _Float16* ctx_h   = (_Float16*)alloc((size_t)B * L * 2);
    _Float16* WkT_top = (_Float16*)alloc((size_t)H4 * L * 2);
    _Float16* WkT_bot = (_Float16*)alloc((size_t)H4 * L * 2);
    _Float16* WrT     = (_Float16*)alloc((size_t)H4 * H * 2);
    _Float16* W1T     = (_Float16*)alloc((size_t)L * L * 2);
    _Float16* W2T_h   = (_Float16*)alloc((size_t)L * H * 2);
    _Float16* W2T_c   = (_Float16*)alloc((size_t)L * H * 2);
    float*    A0      = (float*)alloc((size_t)B * H4 * 4);   // latent@Wk_top + b
    float*    P1      = (float*)alloc((size_t)B * L * 4);    // latent@W1 + b1 + b2
    float*    bsum    = (float*)alloc((size_t)L * 4);
    float*    h_f     = (float*)alloc((size_t)B * H * 4);
    float*    c_f     = (float*)alloc((size_t)B * H * 4);
    float*    zbuf    = (float*)alloc((size_t)B * H4 * 4);
    float*    score   = (float*)alloc((size_t)B * L * 4);

    const dim3 thr256(256);
    auto blocks1d = [](int n) { return dim3((unsigned)((n + 255) / 256)); };

    // ---- one-time prep (runs every call for determinism) ----
    cvt_f16_kernel<<<blocks1d(B * L), thr256, 0, stream>>>(latent, lat_h, B * L);
    cvt_f16_kernel<<<blocks1d(B * H), thr256, 0, stream>>>(h0, h_h, B * H);
    cvt_f16_kernel<<<blocks1d(B * H), thr256, 0, stream>>>(c0, c_h, B * H);
    copy_f32_kernel<<<blocks1d(B * H), thr256, 0, stream>>>(h0, h_f, B * H);
    copy_f32_kernel<<<blocks1d(B * H), thr256, 0, stream>>>(c0, c_f, B * H);
    add_vec_kernel<<<blocks1d(L), thr256, 0, stream>>>(b1, b2, bsum, L);

    const dim3 tthr(32, 8);
    transpose_cvt_kernel<<<dim3(H4 / 32, L / 32), tthr, 0, stream>>>(Wk, WkT_top, L, H4);
    transpose_cvt_kernel<<<dim3(H4 / 32, L / 32), tthr, 0, stream>>>(Wk + (size_t)L * H4, WkT_bot, L, H4);
    transpose_cvt_kernel<<<dim3(H4 / 32, H / 32), tthr, 0, stream>>>(Wr, WrT, H, H4);
    transpose_cvt_kernel<<<dim3(L / 32, L / 32), tthr, 0, stream>>>(W1, W1T, L, L);
    transpose_cvt_kernel<<<dim3(L / 32, H / 32), tthr, 0, stream>>>(W2, W2T_h, H, L);
    transpose_cvt_kernel<<<dim3(L / 32, H / 32), tthr, 0, stream>>>(W2 + (size_t)H * L, W2T_c, H, L);

    // hoisted loop-invariant GEMMs
    gemm_f16_wmma<8><<<dim3(H4 / BN, B / 128), thr256, 0, stream>>>(
        lat_h, WkT_top, nullptr, nullptr, nullptr, bb, A0, B, H4, L, 0);
    gemm_f16_wmma<4><<<dim3(L / BN, B / 64), thr256, 0, stream>>>(
        lat_h, W1T, nullptr, nullptr, nullptr, bsum, P1, B, L, L, 0);

    // initial attention context from (h0, c0)
    gemm_f16_wmma<4><<<dim3(L / BN, B / 64), thr256, 0, stream>>>(
        h_h, W2T_h, c_h, W2T_c, P1, nullptr, score, B, L, H, 1);
    softmax_ctx_kernel<<<dim3(B), thr256, 0, stream>>>(score, latent, ctx_h, L);

    // ---- the T-step scan ----
    for (int t = 0; t < T; ++t) {
        // z = A0 + ctx@Wk_bot + h@Wr
        gemm_f16_wmma<8><<<dim3(H4 / BN, B / 128), thr256, 0, stream>>>(
            ctx_h, WkT_bot, h_h, WrT, A0, nullptr, zbuf, B, H4, H, 0);
        // gates -> h_new, c_new (f32 + f16 copies)
        lstm_gates_kernel<<<blocks1d(B * H), thr256, 0, stream>>>(zbuf, c_f, h_f, c_h, h_h, H);
        // out[:, t] = h_new @ Wmu + bmu
        out_mu_kernel<<<dim3(B), thr256, 0, stream>>>(h_f, Wmu, bmu, out, t, T, H);
        if (t + 1 < T) {
            // score = tanh(P1 + h@W2_h + c@W2_c); ctx = softmax(score)*latent
            gemm_f16_wmma<4><<<dim3(L / BN, B / 64), thr256, 0, stream>>>(
                h_h, W2T_h, c_h, W2T_c, P1, nullptr, score, B, L, H, 1);
            softmax_ctx_kernel<<<dim3(B), thr256, 0, stream>>>(score, latent, ctx_h, L);
        }
    }
    (void)in_sizes; (void)n_in; (void)out_size; (void)ws_size;
}
